// RGCNConv_16587163697548
// MI455X (gfx1250) — matrix operand order repaired and checked by reading, not compiled, not run on hardware
//
#include <hip/hip_runtime.h>

// ---------------------------------------------------------------------------
// RGCNConv forward for MI455X (gfx1250, wave32, WMMA).
// out = x@Wself + sum_r scatter(dst_r, gather(src_r, x@Wr) * invdeg_r[dst])
// GEMMs in bf16 via v_wmma_f32_16x16x32_bf16; scatter via global_atomic_add_f32.
// ---------------------------------------------------------------------------

typedef __attribute__((ext_vector_type(16))) __bf16 v16bf;
typedef __attribute__((ext_vector_type(8)))  __bf16 v8bf;
typedef __attribute__((ext_vector_type(8)))  float  v8f;

#define IN_CH 128   // K
#define OUT_CH 128  // N

// --------------------------- conversion kernels ----------------------------

__global__ void __launch_bounds__(256)
k_conv_x(const float* __restrict__ x, __bf16* __restrict__ xb, int total) {
    int i = blockIdx.x * blockDim.x + threadIdx.x;
    if (i < total) xb[i] = (__bf16)x[i];
}

// Build bf16 transposed weights: wtb[mat][n][k] = W[mat][k][n].
// mat 0 = self_weight, mats 1..R = rel_weights.
__global__ void __launch_bounds__(256)
k_conv_w(const float* __restrict__ wSelf, const float* __restrict__ wRel,
         __bf16* __restrict__ wtb, int total) {
    int i = blockIdx.x * blockDim.x + threadIdx.x;
    if (i >= total) return;
    int mat = i >> 14;          // / (128*128)
    int rem = i & 16383;
    int n   = rem >> 7;         // output row  (N index)
    int k   = rem & 127;        // output col  (K index)
    float v = (mat == 0) ? wSelf[k * OUT_CH + n]
                         : wRel[(size_t)(mat - 1) * (IN_CH * OUT_CH) + k * OUT_CH + n];
    wtb[i] = (__bf16)v;
}

// ------------------------------- WMMA GEMM ---------------------------------
// C[M,128] = A[M,128] (bf16, row major) * Bt^T  where Bt[n][k] (bf16).
// One 16x16 tile per wave; 8 waves per block cover N=128; grid.x = M/16.

__global__ void __launch_bounds__(256)
k_gemm_bf16(const __bf16* __restrict__ A, const __bf16* __restrict__ Bt,
            float* __restrict__ C, int M) {
    const int lane    = threadIdx.x & 31;
    const int wave    = threadIdx.x >> 5;
    const int mBase   = blockIdx.x * 16;
    const int nBase   = wave * 16;
    if (mBase >= M) return;

    const int halfSel = lane >> 4;       // 0: lanes 0-15, 1: lanes 16-31
    const int rowA    = mBase + (lane & 15);
    const int rowB    = nBase + (lane & 15);

    // A fragment: lane<16 holds K = k0..k0+15, lane>=16 holds K = k0+16..k0+31
    const __bf16* aRow = A  + (size_t)rowA * IN_CH + halfSel * 16;
    // B fragment: lane<16 holds K = k0..k0+15, lane>=16 holds K = k0+16..k0+31
    const __bf16* bRow = Bt + (size_t)rowB * IN_CH + halfSel * 16;

    v8f c = {};
#pragma unroll
    for (int k0 = 0; k0 < IN_CH; k0 += 32) {
        v16bf a = *(const v16bf*)(aRow + k0);
        v16bf b = *(const v16bf*)(bRow + k0);
        c = __builtin_amdgcn_wmma_f32_16x16x32_bf16(
                /*neg_a=*/false, a, /*neg_b=*/false, b,
                /*c_mod=*/(short)0, c, /*reuse_a=*/false, /*reuse_b=*/false);
    }

    // C layout: VGPR r -> M = r + halfSel*8 (lanes mod 16 give N)
    float* cBase = C + (size_t)(mBase + halfSel * 8) * OUT_CH + nBase + (lane & 15);
#pragma unroll
    for (int r = 0; r < 8; ++r)
        cBase[(size_t)r * OUT_CH] = c[r];
}

// ------------------------------ degree kernels -----------------------------

__global__ void __launch_bounds__(256)
k_zero(float* __restrict__ p, int n) {
    int i = blockIdx.x * blockDim.x + threadIdx.x;
    if (i < n) p[i] = 0.0f;
}

__global__ void __launch_bounds__(256)
k_count_deg(const int* __restrict__ dst, const unsigned char* __restrict__ mask,
            float* __restrict__ deg, int nEdges) {
    int e = blockIdx.x * blockDim.x + threadIdx.x;
    if (e < nEdges && mask[e])
        unsafeAtomicAdd(&deg[dst[e]], 1.0f);
}

__global__ void __launch_bounds__(256)
k_invert_deg(float* __restrict__ deg, int n) {
    int i = blockIdx.x * blockDim.x + threadIdx.x;
    if (i < n) {
        float d = deg[i];
        deg[i] = (d > 0.0f) ? (1.0f / d) : 0.0f;
    }
}

// ------------------------------ fused scatter ------------------------------
// 32 lanes per edge, float4 per lane: out[dst] += h[src] * invdeg[dst]

__global__ void __launch_bounds__(256)
k_scatter(const float* __restrict__ H, const int* __restrict__ src,
          const int* __restrict__ dst, const unsigned char* __restrict__ mask,
          const float* __restrict__ invdeg, float* __restrict__ out, int nEdges) {
    int t = blockIdx.x * blockDim.x + threadIdx.x;
    int e = t >> 5;
    if (e >= nEdges) return;
    if (!mask[e]) return;
    int lane = t & 31;
    int s = src[e];
    int d = dst[e];
    float w = invdeg[d];
    const float4 v = *(const float4*)(H + (size_t)s * OUT_CH + lane * 4);
    float* o = out + (size_t)d * OUT_CH + lane * 4;
    unsafeAtomicAdd(o + 0, v.x * w);
    unsafeAtomicAdd(o + 1, v.y * w);
    unsafeAtomicAdd(o + 2, v.z * w);
    unsafeAtomicAdd(o + 3, v.w * w);
}

// --------------------------------- launch ----------------------------------

extern "C" void kernel_launch(void* const* d_in, const int* in_sizes, int n_in,
                              void* d_out, int out_size, void* d_ws, size_t ws_size,
                              hipStream_t stream) {
    const float*         x     = (const float*)d_in[0];
    const int*           eidx  = (const int*)d_in[1];           // [R,2,E]
    const unsigned char* masks = (const unsigned char*)d_in[2]; // [R,E] bool
    const float*         wSelf = (const float*)d_in[3];         // [128,128]
    const float*         wRel  = (const float*)d_in[4];         // [R,128,128]
    float*               out   = (float*)d_out;                 // [N,128]

    const int N = in_sizes[0] / IN_CH;                // 100000
    const int R = in_sizes[4] / (IN_CH * OUT_CH);     // 16
    const int E = in_sizes[1] / (2 * R);              // 200000

    // ---- workspace carve-out (all 256B aligned) ----
    char*  wsb = (char*)d_ws;
    size_t off = 0;
    __bf16* xb  = (__bf16*)(wsb + off);  off += (size_t)N * IN_CH * sizeof(__bf16);
    off = (off + 255) & ~(size_t)255;
    __bf16* wtb = (__bf16*)(wsb + off);  off += (size_t)(R + 1) * IN_CH * OUT_CH * sizeof(__bf16);
    off = (off + 255) & ~(size_t)255;
    float*  h   = (float*)(wsb + off);   off += (size_t)N * OUT_CH * sizeof(float);
    off = (off + 255) & ~(size_t)255;
    float*  deg = (float*)(wsb + off);   off += (size_t)N * sizeof(float);
    (void)ws_size;

    const int BLK = 256;
    // 1. convert x -> bf16, build transposed bf16 weights
    {
        int totX = N * IN_CH;
        k_conv_x<<<(totX + BLK - 1) / BLK, BLK, 0, stream>>>(x, xb, totX);
        int totW = (R + 1) * IN_CH * OUT_CH;
        k_conv_w<<<(totW + BLK - 1) / BLK, BLK, 0, stream>>>(wSelf, wRel, wtb, totW);
    }

    const int gemmGrid = (N + 15) / 16;   // 6250 blocks, 8 waves each

    // 2. self-loop transform straight into out (fully overwrites d_out)
    k_gemm_bf16<<<gemmGrid, BLK, 0, stream>>>(xb, wtb, out, N);

    // 3. per-relation: deg -> invdeg -> h = x@Wr -> fused normalize+scatter
    for (int r = 0; r < R; ++r) {
        const int*           srcR  = eidx + (size_t)r * 2 * E;
        const int*           dstR  = srcR + E;
        const unsigned char* maskR = masks + (size_t)r * E;
        const __bf16*        wtR   = wtb + (size_t)(r + 1) * IN_CH * OUT_CH;

        k_zero      <<<(N + BLK - 1) / BLK, BLK, 0, stream>>>(deg, N);
        k_count_deg <<<(E + BLK - 1) / BLK, BLK, 0, stream>>>(dstR, maskR, deg, E);
        k_invert_deg<<<(N + BLK - 1) / BLK, BLK, 0, stream>>>(deg, N);
        k_gemm_bf16 <<<gemmGrid, BLK, 0, stream>>>(xb, wtR, h, N);

        long long scatterThreads = (long long)E * 32;
        int scatterGrid = (int)((scatterThreads + BLK - 1) / BLK);
        k_scatter<<<scatterGrid, BLK, 0, stream>>>(h, srcR, dstR, maskR, deg, out, E);
    }
}
